// DynamicConv1D_74234214744818
// MI455X (gfx1250) — matrix operand order repaired and checked by reading, not compiled, not run on hardware
//
#include <hip/hip_runtime.h>

typedef __attribute__((ext_vector_type(16))) _Float16 v16h;
typedef __attribute__((ext_vector_type(8)))  float    v8f;

#define IN_CH   32
#define OUT_CH  32
#define KTAP    7
#define PADT    3
#define TLEN    4096
#define NBATCH  4

#define WAVES          4
#define TILES_PER_WAVE 4
#define T_PER_WG       256                    // 16 tiles of 16
#define XT_ROWS        (T_PER_WG + KTAP - 1)  // 262 staged time points
#define XC_STRIDE      266                    // padded to kill bank conflicts

// LDS byte offsets
#define AW_ELEMS   (2 * 7 * 7 * 32 * 16)              // 50176 f16 (A fragments)
#define OFF_XT     (AW_ELEMS * 2)                     // 100352
#define OFF_XC     (OFF_XT + XT_ROWS * 32 * 2)        // +16768 = 117120
#define OFF_BIAS   (OFF_XC + IN_CH * XC_STRIDE * 4)   // +34048 = 151168
#define OFF_YACC   (OFF_BIAS + OUT_CH * KTAP * 4)     // +896   = 152064
#define SMEM_BYTES (OFF_YACC + T_PER_WG * 4)          // +1024  = 153088

__global__ __launch_bounds__(128)
void dynconv1d_fused(
    const float* __restrict__ x, const float* __restrict__ W,
    const float* __restrict__ bias, float* __restrict__ y)
{
    extern __shared__ char smem[];
    _Float16* AW   = (_Float16*)smem;                 // [2][7 tap][7 kp][32 lanes][16 f16]
    _Float16* xT   = (_Float16*)(smem + OFF_XT);      // [262][32] f16 (t-major)
    float*    xC   = (float*)   (smem + OFF_XC);      // [32][266] f32 (c-major)
    float*    bsh  = (float*)   (smem + OFF_BIAS);    // [32][7]
    float*    yacc = (float*)   (smem + OFF_YACC);    // [256] per-WG partial y

    const int tid = threadIdx.x;
    const int bid = blockIdx.x;
    const int b  = bid >> 9;          // / (32*16)
    const int o  = (bid >> 4) & 31;
    const int tb = bid & 15;
    const int t0 = tb * T_PER_WG;

    // ---- stage weights for this o: coalesced global read, shuffle into
    //      WMMA A-fragment lane layout (contiguous k = 16*h + e, matching B) ----
    const float* Wo = W + (size_t)o * 32 * 7 * 224;  // rows (i,k), inner c*7+k'
    for (int s = tid; s < 50176; s += 128) {
        int i  = s / 1568;
        int r  = s - i * 1568;
        int k  = r / 224;  r -= k * 224;
        int c  = r / 7;
        int kp = r - c * 7;
        float wv = Wo[s];
        int ihalf = i >> 4, il = i & 15;
        int h, v;
        h = c >> 4; v = (c & 15) >> 1;
        int lane = h * 16 + il;
        int e    = 2 * v + (c & 1);
        AW[(((ihalf * 7 + k) * 7 + kp) * 32 + lane) * 16 + e] = (_Float16)wv;
    }
    // ---- stage x slab [t0-3, t0+258]: f16 t-major (B operands) + f32 c-major ----
    const float* xb = x + (size_t)b * IN_CH * TLEN;
    for (int s = tid; s < IN_CH * XT_ROWS; s += 128) {
        int c  = s / XT_ROWS;
        int lt = s - c * XT_ROWS;
        int t  = t0 - PADT + lt;
        float v = (t >= 0 && t < TLEN) ? xb[c * TLEN + t] : 0.0f;
        xC[c * XC_STRIDE + lt] = v;
        xT[lt * 32 + c] = (_Float16)v;
    }
    for (int s = tid; s < OUT_CH * KTAP; s += 128) bsh[s] = bias[o * 224 + s];

    __syncthreads();

    const int wave = tid >> 5;
    const int lane = tid & 31;
    const int nn   = lane & 15;   // N = local t
    const int hh   = lane >> 4;
    const float inv_sqrt7 = 0.3779644730f;

    #pragma unroll
    for (int ihalf = 0; ihalf < 2; ++ihalf) {
        // ---- weight-stationary: 49 A fragments for this i-half -> 392 VGPRs,
        //      loaded ONCE and reused for all 4 t-tiles ----
        v16h areg[KTAP][KTAP];
        #pragma unroll
        for (int k = 0; k < KTAP; ++k)
            #pragma unroll
            for (int kp = 0; kp < KTAP; ++kp)
                areg[k][kp] = *(const v16h*)(
                    AW + ((((ihalf * 7 + k) * 7 + kp) * 32 + lane) << 4));

        // bias fragments (row-constant across N), reused for all tiles
        v8f bias_frag[KTAP];
        #pragma unroll
        for (int k = 0; k < KTAP; ++k)
            #pragma unroll
            for (int j = 0; j < 8; ++j)
                bias_frag[k][j] = bsh[(ihalf * 16 + j + 8 * hh) * 7 + k];

        #pragma unroll 1
        for (int tl = 0; tl < TILES_PER_WAVE; ++tl) {
            const int ltbase = (wave * TILES_PER_WAVE + tl) * 16;

            v8f acc[KTAP];
            #pragma unroll
            for (int k = 0; k < KTAP; ++k) acc[k] = bias_frag[k];

            // K-chunk outer: one B fragment (time-shifted x tile) feeds
            // 7 consecutive WMMAs; acc RAW distance = 7 instructions
            #pragma unroll
            for (int kp = 0; kp < KTAP; ++kp) {
                const v16h bf = *(const v16h*)(
                    xT + (ltbase + nn + kp) * 32 + hh * 16);
                #pragma unroll
                for (int k = 0; k < KTAP; ++k)
                    acc[k] = __builtin_amdgcn_wmma_f32_16x16x32_f16(
                        false, areg[k][kp], false, bf, (short)0, acc[k],
                        false, false);
            }

            // per-element softmax over the 7 taps + weighted gather of x
            float contrib = 0.0f;
            #pragma unroll
            for (int j = 0; j < 8; ++j) {
                float m = acc[0][j];
                #pragma unroll
                for (int k = 1; k < KTAP; ++k) m = fmaxf(m, acc[k][j]);
                float e[KTAP], es = 0.0f;
                #pragma unroll
                for (int k = 0; k < KTAP; ++k) {
                    e[k] = __expf((acc[k][j] - m) * inv_sqrt7);
                    es += e[k];
                }
                const float inv = __builtin_amdgcn_rcpf(es);
                const int   i   = ihalf * 16 + j + 8 * hh;
                const float* xrow = xC + i * XC_STRIDE + ltbase + nn;
                float dot = 0.0f;
                #pragma unroll
                for (int k = 0; k < KTAP; ++k) dot += e[k] * xrow[k];
                contrib += dot * inv;
            }
            contrib += __shfl_xor(contrib, 16, 32);  // fold the two M-halves

            // bridge the two i-half passes through per-wave LDS partials
            if (ihalf == 0) {
                if (lane < 16) yacc[ltbase + nn] = contrib;
            } else {
                if (lane < 16)
                    y[((size_t)b * OUT_CH + o) * TLEN + t0 + ltbase + nn] =
                        contrib + yacc[ltbase + nn];
            }
        }
    }
}

extern "C" void kernel_launch(void* const* d_in, const int* in_sizes, int n_in,
                              void* d_out, int out_size, void* d_ws, size_t ws_size,
                              hipStream_t stream) {
    (void)in_sizes; (void)n_in; (void)d_ws; (void)ws_size; (void)out_size;
    const float* x  = (const float*)d_in[0];
    const float* W  = (const float*)d_in[1];
    const float* bb = (const float*)d_in[2];
    float* y = (float*)d_out;

    hipFuncSetAttribute((const void*)dynconv1d_fused,
                        hipFuncAttributeMaxDynamicSharedMemorySize, SMEM_BYTES);

    dim3 grid(NBATCH * OUT_CH * (TLEN / T_PER_WG));  // 4*32*16 = 2048
    dynconv1d_fused<<<grid, WAVES * 32, SMEM_BYTES, stream>>>(x, W, bb, y);
}
